// LSTM_74131135529259
// MI455X (gfx1250) — compile-verified, matrix operand order checked by reading, not run on hardware
//
#include <hip/hip_runtime.h>
#include <hip/hip_bf16.h>
#include <math.h>

typedef __attribute__((ext_vector_type(16))) __bf16 v16bf;
typedef __attribute__((ext_vector_type(8)))  float  v8f;

#define Tn 256
#define Bn 128
#define Dn 512
#define Hn 512
#define NG 2048              // 4*H
#define Mrows (Tn*Bn)        // 32768
#define GMT 4                // M-tiles per wave in big GEMM
#define GNT 2                // N-tiles per wave in big GEMM

static __device__ __forceinline__ unsigned short f32_to_bf16_bits(float f) {
    unsigned u = __builtin_bit_cast(unsigned, f);
    unsigned r = u + 0x7FFFu + ((u >> 16) & 1u);   // round-to-nearest-even
    return (unsigned short)(r >> 16);
}
static __device__ __forceinline__ float bf16_bits_to_f32(unsigned short b) {
    unsigned u = ((unsigned)b) << 16;
    return __builtin_bit_cast(float, u);
}
// 16-bit A-matrix 16x32 K placement (ISA 7.12.2): vgpr v, lane-half h
static __device__ __forceinline__ int a_koff(int v, int half) {
    return (v < 4) ? (half * 8 + 2 * v) : (16 + half * 8 + 2 * (v - 4));
}

// ---------------- prologue kernels ----------------

__global__ void cvt_f32_to_bf16(const float* __restrict__ in,
                                unsigned short* __restrict__ out, int n) {
    int i = blockIdx.x * blockDim.x + threadIdx.x;
    if (i < n) out[i] = f32_to_bf16_bits(in[i]);
}

// Pack 4 gate weight matrices [512,512] (row=K, col=N) into WMMA B-fragment
// order: tile(gate,nt,kt) -> 32 lanes x 16 bf16, lane holds col n=lane%16,
// K pattern mirrors the 16-bit A layout. Flat out index == thread index.
__global__ void pack_w(const float* __restrict__ W0, const float* __restrict__ W1,
                       const float* __restrict__ W2, const float* __restrict__ W3,
                       unsigned short* __restrict__ Wp) {
    int idx  = blockIdx.x * blockDim.x + threadIdx.x;   // < 4*32*16*512
    int e    = idx & 15;
    int lane = (idx >> 4) & 31;
    int kt   = (idx >> 9) & 15;
    int nt   = (idx >> 13) & 31;
    int gate = idx >> 18;
    int v = e >> 1, odd = e & 1;
    int k = kt * 32 + a_koff(v, lane >> 4) + odd;
    int n = nt * 16 + (lane & 15);
    const float* W = (gate == 0) ? W0 : (gate == 1) ? W1 : (gate == 2) ? W2 : W3;
    Wp[idx] = f32_to_bf16_bits(W[(size_t)k * Hn + n]);
}

__global__ void pack_b(const float* __restrict__ b0, const float* __restrict__ b1,
                       const float* __restrict__ b2, const float* __restrict__ b3,
                       float* __restrict__ bcat) {
    int i = blockIdx.x * blockDim.x + threadIdx.x;      // < 2048
    const float* b = (i < 512) ? b0 : (i < 1024) ? b1 : (i < 1536) ? b2 : b3;
    bcat[i] = b[i & 511];
}

__global__ void zero_state(unsigned short* __restrict__ h0, float* __restrict__ s) {
    int i = blockIdx.x * blockDim.x + threadIdx.x;      // < 128*512
    h0[i] = 0;
    s[i] = 0.0f;
}

// ---------------- big input-projection GEMM:  Xg = X*Wx + b ----------------
// Register-tiled: each wave owns a 4x2 block of 16x16 tiles. Per K-step,
// 4 A-frags + 2 B-frags feed 8 WMMAs (operand reuse across the block).
// Output stored as bf16 C-fragments (tile-major, lane*8 contiguous).
__global__ void __launch_bounds__(256)
xg_gemm(const unsigned short* __restrict__ Xb,     // [32768,512] bf16 row-major
        const unsigned short* __restrict__ Wxp,    // packed B fragments
        const float* __restrict__ bcat,            // [2048]
        unsigned short* __restrict__ Xgp) {        // [2048 mt][128 ntG][256]
    int wid  = blockIdx.x * 8 + (threadIdx.x >> 5);
    int lane = threadIdx.x & 31;
    int mg   = wid >> 6;          // 0..511   M-tile group (4 tiles)
    int ng   = wid & 63;          // 0..63    N-tile group (2 tiles)
    int half = lane >> 4;

    const unsigned short* arow[GMT];
#pragma unroll
    for (int i = 0; i < GMT; ++i)
        arow[i] = Xb + (size_t)((mg * GMT + i) * 16 + (lane & 15)) * Dn;

    v8f acc[GMT][GNT];
    v8f zero = {};
#pragma unroll
    for (int i = 0; i < GMT; ++i)
#pragma unroll
        for (int j = 0; j < GNT; ++j) acc[i][j] = zero;

    for (int kt = 0; kt < Dn / 32; ++kt) {
        v16bf Af[GMT];
#pragma unroll
        for (int i = 0; i < GMT; ++i) {
            union { unsigned u[8]; v16bf v; } A;
#pragma unroll
            for (int v = 0; v < 8; ++v)
                A.u[v] = *(const unsigned*)(arow[i] + kt * 32 + a_koff(v, half));
            Af[i] = A.v;
        }
        v16bf Bf[GNT];
#pragma unroll
        for (int j = 0; j < GNT; ++j)
            Bf[j] = *(const v16bf*)(Wxp + ((size_t)(ng * GNT + j) * 16 + kt) * 512 + lane * 16);
#pragma unroll
        for (int i = 0; i < GMT; ++i)
#pragma unroll
            for (int j = 0; j < GNT; ++j)
                acc[i][j] = __builtin_amdgcn_wmma_f32_16x16x32_bf16(
                    false, Af[i], false, Bf[j], (short)0, acc[i][j], false, false);
    }

#pragma unroll
    for (int j = 0; j < GNT; ++j) {
        int ntG = ng * GNT + j;
        float bias = bcat[ntG * 16 + (lane & 15)];
#pragma unroll
        for (int i = 0; i < GMT; ++i) {
            int mt = mg * GMT + i;
            union { unsigned short s[8]; uint4 q; } O;
#pragma unroll
            for (int r = 0; r < 8; ++r) O.s[r] = f32_to_bf16_bits(acc[i][j][r] + bias);
            *(uint4*)(Xgp + ((size_t)mt * 128 + ntG) * 256 + lane * 8) = O.q;
        }
    }
}

// ---------------- recurrent step: g = Xg[t] + h_prev*Wh ; gates ----------------
// 256 waves, each owns one (mt,nt) tile for ALL 4 gates -> gate math in regs;
// each A-fragment is reused by 4 gate WMMAs.
__global__ void __launch_bounds__(256)
lstm_step(int t,
          const unsigned short* __restrict__ Xgp,
          const unsigned short* __restrict__ Whp,
          const unsigned short* __restrict__ h_in,   // [128,512] bf16
          unsigned short* __restrict__ h_out,        // [128,512] bf16
          float* __restrict__ s_ws,                  // [128,512] f32, in place
          float* __restrict__ Y) {                   // [256,128,512] f32
    int wid  = blockIdx.x * 8 + (threadIdx.x >> 5);  // 0..255
    int lane = threadIdx.x & 31;
    int mt   = wid >> 5;        // 0..7
    int nt   = wid & 31;        // 0..31
    int half = lane >> 4;
    int n16  = lane & 15;

    // init accumulators from precomputed input projection (bf16 C-fragments)
    v8f acc[4];
#pragma unroll
    for (int g = 0; g < 4; ++g) {
        const unsigned short* xg =
            Xgp + (((size_t)(t * 8 + mt) * 128) + g * 32 + nt) * 256 + lane * 8;
        union { uint4 q; unsigned short s[8]; } I;
        I.q = *(const uint4*)xg;
#pragma unroll
        for (int r = 0; r < 8; ++r) acc[g][r] = bf16_bits_to_f32(I.s[r]);
    }

    const unsigned short* arow = h_in + (size_t)(mt * 16 + n16) * Hn;
    for (int kt = 0; kt < Hn / 32; ++kt) {
        union { unsigned u[8]; v16bf v; } A;
#pragma unroll
        for (int v = 0; v < 8; ++v)
            A.u[v] = *(const unsigned*)(arow + kt * 32 + a_koff(v, half));
#pragma unroll
        for (int g = 0; g < 4; ++g) {
            v16bf Bf = *(const v16bf*)(Whp + ((size_t)(g * 32 + nt) * 16 + kt) * 512 + lane * 16);
            acc[g] = __builtin_amdgcn_wmma_f32_16x16x32_bf16(false, A.v, false, Bf,
                                                             (short)0, acc[g], false, false);
        }
    }

    int n = nt * 16 + n16;
#pragma unroll
    for (int r = 0; r < 8; ++r) {
        int m = mt * 16 + r + 8 * half;        // C-layout: VGPR r, lane-half
        size_t idx = (size_t)m * Hn + n;
        float a  = tanhf(acc[0][r]);
        float ig = 1.0f / (1.0f + __expf(-acc[1][r]));
        float fg = 1.0f / (1.0f + __expf(-acc[2][r]));
        float og = 1.0f / (1.0f + __expf(-acc[3][r]));
        float sn = a * ig + s_ws[idx] * fg;
        s_ws[idx] = sn;
        float h = tanhf(sn) * og;
        Y[(size_t)t * (Bn * Hn) + idx] = h;
        h_out[idx] = f32_to_bf16_bits(h);
    }
}

// ---------------- host ----------------

extern "C" void kernel_launch(void* const* d_in, const int* in_sizes, int n_in,
                              void* d_out, int out_size, void* d_ws, size_t ws_size,
                              hipStream_t stream) {
    const float* X   = (const float*)d_in[0];
    const float* Wax = (const float*)d_in[1];
    const float* Wix = (const float*)d_in[2];
    const float* Wfx = (const float*)d_in[3];
    const float* Wox = (const float*)d_in[4];
    const float* Wah = (const float*)d_in[5];
    const float* Wih = (const float*)d_in[6];
    const float* Wfh = (const float*)d_in[7];
    const float* Woh = (const float*)d_in[8];
    const float* ba  = (const float*)d_in[9];
    const float* bi  = (const float*)d_in[10];
    const float* bf  = (const float*)d_in[11];
    const float* bo  = (const float*)d_in[12];
    (void)in_sizes; (void)n_in; (void)out_size; (void)ws_size;

    char* ws = (char*)d_ws;
    size_t off = 0;
    auto carve = [&](size_t bytes) {
        char* p = ws + off;
        off = (off + bytes + 255) & ~(size_t)255;
        return p;
    };
    unsigned short* Xb  = (unsigned short*)carve((size_t)Mrows * Dn * 2);   // 32 MB
    unsigned short* Wxp = (unsigned short*)carve((size_t)4 * Hn * Hn * 2);  // 2 MB
    unsigned short* Whp = (unsigned short*)carve((size_t)4 * Hn * Hn * 2);  // 2 MB
    float*          bc  = (float*)carve((size_t)NG * 4);
    unsigned short* Xgp = (unsigned short*)carve((size_t)Mrows * NG * 2);   // 128 MB
    unsigned short* h0  = (unsigned short*)carve((size_t)Bn * Hn * 2);
    unsigned short* h1  = (unsigned short*)carve((size_t)Bn * Hn * 2);
    float*          sws = (float*)carve((size_t)Bn * Hn * 4);

    // prologue: convert / pack / zero
    cvt_f32_to_bf16<<<(Mrows * Dn) / 256, 256, 0, stream>>>(X, Xb, Mrows * Dn);
    pack_w<<<(4 * Hn * Hn) / 256, 256, 0, stream>>>(Wax, Wix, Wfx, Wox, Wxp);
    pack_w<<<(4 * Hn * Hn) / 256, 256, 0, stream>>>(Wah, Wih, Wfh, Woh, Whp);
    pack_b<<<NG / 256, 256, 0, stream>>>(ba, bi, bf, bo, bc);
    zero_state<<<(Bn * Hn) / 256, 256, 0, stream>>>(h0, sws);

    // big input-projection GEMM: (2048/4 M-groups) x (128/2 N-groups) waves
    xg_gemm<<<((2048 / GMT) * (128 / GNT)) / 8, 256, 0, stream>>>(Xb, Wxp, bc, Xgp);

    // sequential recurrence: one kernel per timestep, ping-pong h buffers
    float* Y = (float*)d_out;
    for (int t = 0; t < Tn; ++t) {
        const unsigned short* hi = (t & 1) ? h1 : h0;
        unsigned short*       ho = (t & 1) ? h0 : h1;
        lstm_step<<<32, 256, 0, stream>>>(t, Xgp, Whp, hi, ho, sws, Y);
    }
}
